// PatchMatch_51290499449250
// MI455X (gfx1250) — matrix operand order repaired and checked by reading, not compiled, not run on hardware
//
#include <hip/hip_runtime.h>
#include <hip/hip_bf16.h>
#include <math.h>

typedef __attribute__((ext_vector_type(16))) _Float16 v16h;
typedef __attribute__((ext_vector_type(8)))  float    v8f;

#define C_CH 32
#define H_IMG 128
#define W_IMG 256
#define S_SAMP 12
#define NTILE 16              // W/16 wL tiles per row
#define NOFF 10               // wR-tile offsets o = (wR>>4)-(wL>>4)+8 in [0,10)
#define CORR_SCALE (7.0f / 32.0f)   // TEMPERATURE / C

// corr layout (tile-major, branch-free WMMA stores):
//   corr[((h*NTILE + t)*NOFF + o)*256 + M*16 + N]
//     = (T/C) * sum_c left[c,h, 16t+M] * right[c,h, 16(t+o-8)+N]
#define CORR_FLOATS ((size_t)H_IMG * NTILE * NOFF * 256)

// ---------------------------------------------------------------------------
// Phase 1: row correlation via WMMA. One block per image row h; 8 waves;
// each wave owns 2 wL-tiles and sweeps the in-image wR tiles o in [8-t .. 9].
// K = C = 32 -> one v_wmma_f32_16x16x32_f16 per 16x16 output tile; the full
// tile is stored unpredicated (coalesced b32 stores, no exec juggling).
// ---------------------------------------------------------------------------
__global__ __launch_bounds__(256) void corr_tile_kernel(
    const float* __restrict__ left, const float* __restrict__ right,
    float* __restrict__ corr) {
  // LDS staging of one row of left/right as f16, w-major (fragment's 8
  // consecutive K values = 16 contiguous bytes -> ds_load_b128).
  __shared__ _Float16 ldsL[W_IMG * C_CH];  // 16 KB
  __shared__ _Float16 ldsR[W_IMG * C_CH];  // 16 KB

  const int h   = blockIdx.x;
  const int tid = threadIdx.x;             // 256 threads: one per w
  for (int k = 0; k < C_CH; ++k) {
    ldsL[tid * C_CH + k] = (_Float16)left [k * (H_IMG * W_IMG) + h * W_IMG + tid];
    ldsR[tid * C_CH + k] = (_Float16)right[k * (H_IMG * W_IMG) + h * W_IMG + tid];
  }
  __syncthreads();

  const int wave = tid >> 5;      // 0..7
  const int lane = tid & 31;
  const int half = lane >> 4;     // lane group 0/1
  const int m16  = lane & 15;

  for (int ti = 0; ti < 2; ++ti) {
    const int t  = wave * 2 + ti; // wL tile index 0..15
    const int t0 = t * 16;

    // ---- A fragment: A[m][k] = left[k, t0+m]  (16x32 f16 layout) ----
    v16h a;
    {
      const int wA = t0 + m16;
#pragma unroll
      for (int j = 0; j < 16; ++j) {
        const int K = (j < 8) ? (half * 8 + j) : (16 + half * 8 + (j - 8));
        a[j] = ldsL[wA * C_CH + K];
      }
    }

    // o in [0,10): rbase = t0 + (o-8)*16; keep only in-image tiles.
    const int olo = (t >= 8)  ? 0 : (8 - t);
    const int ohi = (t <= 14) ? 9 : 8;
    for (int o = olo; o <= ohi; ++o) {
      const int rbase = t0 + (o - 8) * 16;

      // ---- B fragment: B[k][n] = right[k, rbase+n]  (32x16 f16 layout) ----
      v16h b;
      {
        const int wB = rbase + m16;
#pragma unroll
        for (int j = 0; j < 16; ++j) {
          const int K = (j < 8) ? (half * 8 + j) : (16 + half * 8 + (j - 8));
          b[j] = ldsR[wB * C_CH + K];
        }
      }

      v8f c = {};
      c = __builtin_amdgcn_wmma_f32_16x16x32_f16(
          /*neg_a=*/false, a, /*neg_b=*/false, b,
          /*c_mod=*/(short)0, c, /*reuse_a=*/false, /*reuse_b=*/false);

      // ---- branch-free tile store: D VGPR r -> M = r + 8*half, N = m16 ----
      float* tileBase =
          corr + (((size_t)h * NTILE + t) * NOFF + o) * 256 + half * 128 + m16;
#pragma unroll
      for (int r = 0; r < 8; ++r) {
        tileBase[r * 16] = c[r] * CORR_SCALE;   // coalesced across lanes
      }
    }
  }
}

// ---------------------------------------------------------------------------
// Phase 2: one PatchMatch round (propagate + evaluate) as L2 tile lookups.
// Grid: one block per row h, one thread per w.
// Lookup of corr(wL=w, wR=x):  t=w>>4, m=w&15, o=(x>>4)-t+8, n=x&15.
// Validity: x in [0,W) implies o in [3,9] (since d < 72), so the x-range
// check alone guarantees we only read tiles phase 1 wrote.
// ---------------------------------------------------------------------------
__global__ __launch_bounds__(256) void pm_round_kernel(
    const float* __restrict__ corr, const float* __restrict__ minDisp,
    const float* __restrict__ maxDisp, const float* __restrict__ srcNoise,
    float* __restrict__ dstNoise, float* __restrict__ dispOut,
    int horizontal, int writeOut) {
  const int w   = threadIdx.x;
  const int h   = blockIdx.x;
  const int pix = h * W_IMG + w;
  const int t   = w >> 4;
  const int m   = w & 15;

  // base pointer for this pixel's row of tiles, pre-offset by M*16 and by
  // the (8 - t) tile-offset bias folded into o via (x>>4)<<8.
  const float* rowBase =
      corr + (((size_t)h * NTILE + t) * NOFF + (8 - t)) * 256 + m * 16;

  const float mn       = fmaxf(minDisp[pix], 0.0f);
  const float mx       = fmaxf(maxDisp[pix], 0.0f);
  const float search   = mx - mn;
  const float interval = 1.0f / (float)(S_SAMP + 1);

  for (int s = 0; s < S_SAMP; ++s) {
    const float imin = mn + search * interval * (float)(s + 1);

    // propagate: 3 taps of noise (zero-padded shifts along W or H)
    float np[3];
#pragma unroll
    for (int f = 0; f < 3; ++f) {
      if (horizontal) {
        const int ww = w + f - 1;
        np[f] = (ww >= 0 && ww < W_IMG)
                    ? srcNoise[((size_t)s * H_IMG + h) * W_IMG + ww] : 0.0f;
      } else {
        const int hh = h + f - 1;
        np[f] = (hh >= 0 && hh < H_IMG)
                    ? srcNoise[((size_t)s * H_IMG + hh) * W_IMG + w] : 0.0f;
      }
    }

    float cost[3], dsamp[3];
#pragma unroll
    for (int f = 0; f < 3; ++f) {
      const float d    = imin + np[f] * search * interval;
      const float xs   = (float)w - d;
      const float x0f  = floorf(xs);
      const float frac = xs - x0f;
      const int   x0i  = (int)x0f;
      const int   x1i  = x0i + 1;
      // tile-major lookup: index = ((x>>4)<<8) + (x&15), relative to rowBase
      const float c0 = (x0i >= 0 && x0i < W_IMG)
                           ? rowBase[((x0i >> 4) << 8) + (x0i & 15)] : 0.0f;
      const float c1 = (x1i >= 0 && x1i < W_IMG)
                           ? rowBase[((x1i >> 4) << 8) + (x1i & 15)] : 0.0f;
      cost[f]  = c0 * (1.0f - frac) + c1 * frac;  // temperature pre-folded
      dsamp[f] = d;
    }

    // softmax over the 3 filter taps
    const float mmax = fmaxf(cost[0], fmaxf(cost[1], cost[2]));
    const float e0 = __expf(cost[0] - mmax);
    const float e1 = __expf(cost[1] - mmax);
    const float e2 = __expf(cost[2] - mmax);
    const float inv = 1.0f / (e0 + e1 + e2);
    const float ds  = (e0 * dsamp[0] + e1 * dsamp[1] + e2 * dsamp[2]) * inv;
    const float dn  = (e0 * np[0]    + e1 * np[1]    + e2 * np[2])    * inv;

    dstNoise[((size_t)s * H_IMG + h) * W_IMG + w] = dn;
    if (writeOut) dispOut[((size_t)s * H_IMG + h) * W_IMG + w] = ds;
  }
}

extern "C" void kernel_launch(void* const* d_in, const int* in_sizes, int n_in,
                              void* d_out, int out_size, void* d_ws, size_t ws_size,
                              hipStream_t stream) {
  (void)in_sizes; (void)n_in; (void)out_size; (void)ws_size;
  const float* left    = (const float*)d_in[0];  // [1,32,128,256]
  const float* right   = (const float*)d_in[1];  // [1,32,128,256]
  const float* minDisp = (const float*)d_in[2];  // [1,1,128,256]
  const float* maxDisp = (const float*)d_in[3];  // [1,1,128,256]
  const float* noise0  = (const float*)d_in[4];  // [1,12,128,256]
  float* out = (float*)d_out;                    // [1,12,128,256]

  // workspace layout: corr tiles (~21 MB) | noiseA (1.5 MB) | noiseB (1.5 MB)
  float* corr   = (float*)d_ws;
  float* noiseA = corr + CORR_FLOATS;
  float* noiseB = noiseA + (size_t)S_SAMP * H_IMG * W_IMG;

  // Phase 1: WMMA row correlation (left/right fixed across all rounds)
  corr_tile_kernel<<<dim3(H_IMG), dim3(256), 0, stream>>>(left, right, corr);

  // Phase 2: 2 iterations x (horizontal, vertical); last round writes output.
  pm_round_kernel<<<dim3(H_IMG), dim3(256), 0, stream>>>(
      corr, minDisp, maxDisp, noise0, noiseA, out, /*horizontal=*/1, /*writeOut=*/0);
  pm_round_kernel<<<dim3(H_IMG), dim3(256), 0, stream>>>(
      corr, minDisp, maxDisp, noiseA, noiseB, out, /*horizontal=*/0, /*writeOut=*/0);
  pm_round_kernel<<<dim3(H_IMG), dim3(256), 0, stream>>>(
      corr, minDisp, maxDisp, noiseB, noiseA, out, /*horizontal=*/1, /*writeOut=*/0);
  pm_round_kernel<<<dim3(H_IMG), dim3(256), 0, stream>>>(
      corr, minDisp, maxDisp, noiseA, noiseB, out, /*horizontal=*/0, /*writeOut=*/1);
}
